// GETLayer_59931973649043
// MI455X (gfx1250) — matrix operand order repaired and checked by reading, not compiled, not run on hardware
//
#include <hip/hip_runtime.h>
#include <hip/hip_bf16.h>

#define HID 128
#define EC  64
#define NH  4
#define HD  32
#define K_NN 8
#define SKK 3
#define NCC 14
#define NRES 768
#define NEDGE (NRES * (K_NN + 1))   // 6912
#define MROW (NRES * NCC)           // 10752

typedef __attribute__((ext_vector_type(16))) _Float16 v16h;
typedef __attribute__((ext_vector_type(8)))  _Float16 v8h;
typedef __attribute__((ext_vector_type(8)))  float    v8f;

// ---------------- WMMA fragment helpers (wave32, gfx1250 layouts) ----------------
// A fragment: 16x32 f16, row-major source. lanes 0-15 row M=lane hold K{0..15},
// lanes 16-31 row M=lane-16 hold K{16..31}. Two contiguous 16B LDS loads.
__device__ __forceinline__ v16h load_a16(const _Float16* A, int lda, int m0, int k0, int lane) {
  int row = m0 + (lane & 15);
  int kLo = k0 + ((lane & 16) ? 16 : 0);
  int kHi = kLo + 8;
  v8h lo = *(const v8h*)(A + row * lda + kLo);
  v8h hi = *(const v8h*)(A + row * lda + kHi);
  return __builtin_shufflevector(lo, hi, 0,1,2,3,4,5,6,7,8,9,10,11,12,13,14,15);
}
// B fragment: 32x16 f16. Source stored N-major (Bt[n][k]) so each lane reads 16
// consecutive K halves: lanes 0-15 K=0..15, lanes 16-31 K=16..31, col N=lane%16.
__device__ __forceinline__ v16h load_b16(const _Float16* Bt, int ldk, int n0, int k0, int lane) {
  int colN = n0 + (lane & 15);
  int kb = k0 + ((lane & 16) ? 16 : 0);
  const v8h* p = (const v8h*)(Bt + colN * ldk + kb);
  v8h lo = p[0], hi = p[1];
  return __builtin_shufflevector(lo, hi, 0,1,2,3,4,5,6,7,8,9,10,11,12,13,14,15);
}
__device__ __forceinline__ v8f wmma16(v16h a, v16h b, v8f c) {
  return __builtin_amdgcn_wmma_f32_16x16x32_f16(false, a, false, b, (short)0, c, false, false);
}
__device__ __forceinline__ float silu_f(float x) { return x / (1.f + __expf(-x)); }

// ---------------- Kernel 1: QKV projection (f16 WMMA, f32 accum) ----------------
struct QkvSmem {
  alignas(16) _Float16 sA[16 * HID];
  alignas(16) _Float16 wt[3 * HID * HID];   // [which][n][k] (N-major for B frags)
};

__global__ void qkv_kernel(const float* __restrict__ res_H,
                           const float* __restrict__ WQ, const float* __restrict__ WK,
                           const float* __restrict__ WV,
                           _Float16* __restrict__ Qh, _Float16* __restrict__ Kh,
                           _Float16* __restrict__ Vh) {
  extern __shared__ char smraw[];
  QkvSmem* s = (QkvSmem*)smraw;
  const int tid = threadIdx.x, wv = tid >> 5, lane = tid & 31;
  const int m0 = blockIdx.x * 16;

  for (int idx = tid; idx < 3 * HID * HID; idx += 256) {
    int which = idx >> 14, rem = idx & 16383, nW = rem >> 7, k = rem & 127;
    const float* W = (which == 0) ? WQ : ((which == 1) ? WK : WV);
    s->wt[idx] = (_Float16)W[k * HID + nW];
  }
  for (int idx = tid; idx < 16 * HID; idx += 256) {
    int r = idx >> 7, k = idx & 127;
    s->sA[idx] = (_Float16)res_H[(size_t)(m0 + r) * HID + k];
  }
  __syncthreads();

  for (int job = wv; job < 24; job += 8) {       // 3 outputs x 8 col-tiles
    int which = job >> 3, nt = job & 7;
    v8f acc = {};
#pragma unroll
    for (int kk = 0; kk < 4; ++kk)
      acc = wmma16(load_a16(s->sA, HID, 0, kk * 32, lane),
                   load_b16(s->wt + which * HID * HID, HID, nt * 16, kk * 32, lane), acc);
    _Float16* O = (which == 0) ? Qh : ((which == 1) ? Kh : Vh);
    int crow = (lane & 16) ? 8 : 0, ccol = nt * 16 + (lane & 15);
#pragma unroll
    for (int r = 0; r < 8; ++r)
      O[(size_t)(m0 + crow + r) * HID + ccol] = (_Float16)acc[r];
  }
}

// ---------------- Kernel 2: per-edge heavy kernel ----------------
struct EdgeSmem {
  alignas(16) _Float16 dist[208 * EC];      // RBF features (rows>=196 zero)
  alignas(16) _Float16 work[208 * HID];     // hidden, then dv
  alignas(16) _Float16 w1t[HID * EC];       // sD_w1^T  [n][k]
  alignas(16) _Float16 w2t[16 * HID];       // sD_w2^T padded to 16 cols
  alignas(16) _Float16 wvt[HID * EC];       // sv_w^T
  alignas(16) _Float16 tiw1t[32 * 96];      // Ti_w1^T
  alignas(16) _Float16 bmi1t[32 * 32];
  alignas(16) _Float16 bmi2t[32 * 32];
  alignas(16) _Float16 q[NH * 16 * 32];     // [h][i][d]
  alignas(16) _Float16 kmat[NH * 16 * 32];  // [h][j][d]  (also B for logits: N=j,K=d)
  alignas(16) _Float16 vt[NH * 32 * 32];    // [h][d][j]  (B for attn@V: N=d,K=j)
  alignas(16) _Float16 attn16[NH * 16 * 32];
  alignas(16) _Float16 av16[NH * 16 * 32];
  alignas(16) _Float16 h2[NH * 16 * 32];
  alignas(16) _Float16 tin[176 * 96];
  alignas(16) _Float16 thid[176 * 32];
  float logit[NH * 16 * 16];
  float attnf[NH * 16 * 16];
  float db[208 * 4];
  float xn[196 * 3];
  float rr[196];
  float amr[NCC], amc[NCC];
  int   topi[NH * NCC * SKK];
  float rowN[NH * 16], rowD[NH * 16];
  float b1[HID], b2[4], svb[HID], tib1[32], tiw2[32], tib2v;
  float bmib1[32], bmib2[32];
  float fbuf[176];
};

__global__ void edge_kernel(
    const float* __restrict__ res_X, const float* __restrict__ atom_mask,
    const int* __restrict__ eidx,
    const float* __restrict__ sD_w1, const float* __restrict__ sD_b1,
    const float* __restrict__ sD_w2, const float* __restrict__ sD_b2,
    const float* __restrict__ sv_w,  const float* __restrict__ sv_b,
    const float* __restrict__ Ti_w1, const float* __restrict__ Ti_b1,
    const float* __restrict__ Ti_w2, const float* __restrict__ Ti_b2,
    const float* __restrict__ bmi_w1, const float* __restrict__ bmi_b1,
    const float* __restrict__ bmi_w2, const float* __restrict__ bmi_b2,
    const _Float16* __restrict__ Qh, const _Float16* __restrict__ Kh,
    const _Float16* __restrict__ Vh,
    float* __restrict__ rij_ws, float* __restrict__ avm_ws, float* __restrict__ fv_ws) {
  extern __shared__ char smraw[];
  EdgeSmem* s = (EdgeSmem*)smraw;
  const int e = blockIdx.x;
  const int tid = threadIdx.x, wv = tid >> 5, lane = tid & 31;
  const int row = eidx[e], col = eidx[NEDGE + e];

  // ---- P0: stage weights (f32 -> f16, N-major), biases, geometry, Q/K/V tiles
  for (int idx = tid; idx < HID * EC; idx += 256) {
    int nW = idx >> 6, k = idx & 63;
    s->w1t[idx] = (_Float16)sD_w1[k * HID + nW];
    s->wvt[idx] = (_Float16)sv_w[k * HID + nW];
  }
  for (int idx = tid; idx < 16 * HID; idx += 256) {
    int nW = idx >> 7, k = idx & 127;
    s->w2t[idx] = (nW < 4) ? (_Float16)sD_w2[k * 4 + nW] : (_Float16)0.f;
  }
  for (int idx = tid; idx < 32 * 96; idx += 256) {
    int nW = idx / 96, k = idx % 96;
    s->tiw1t[idx] = (_Float16)Ti_w1[k * 32 + nW];
  }
  for (int idx = tid; idx < 1024; idx += 256) {
    int nW = idx >> 5, k = idx & 31;
    s->bmi1t[idx] = (_Float16)bmi_w1[k * 32 + nW];
    s->bmi2t[idx] = (_Float16)bmi_w2[k * 32 + nW];
  }
  for (int idx = tid; idx < HID; idx += 256) { s->b1[idx] = sD_b1[idx]; s->svb[idx] = sv_b[idx]; }
  if (tid < 32) { s->tib1[tid] = Ti_b1[tid]; s->tiw2[tid] = Ti_w2[tid];
                  s->bmib1[tid] = bmi_b1[tid]; s->bmib2[tid] = bmi_b2[tid]; }
  if (tid < 4) s->b2[tid] = sD_b2[tid];
  if (tid == 0) s->tib2v = Ti_b2[0];
  if (tid < NCC) { s->amr[tid] = atom_mask[row * NCC + tid]; s->amc[tid] = atom_mask[col * NCC + tid]; }
  if (tid < 196) {
    int i = tid / NCC, j = tid % NCC;
    float dx = res_X[(row * NCC + i) * 3 + 0] - res_X[(col * NCC + j) * 3 + 0];
    float dy = res_X[(row * NCC + i) * 3 + 1] - res_X[(col * NCC + j) * 3 + 1];
    float dz = res_X[(row * NCC + i) * 3 + 2] - res_X[(col * NCC + j) * 3 + 2];
    float r2 = dx * dx + dy * dy + dz * dz;
    s->rr[tid] = sqrtf(r2 + 1e-12f);
    float inv = 1.f / (sqrtf(r2) + 1e-5f);
    s->xn[tid * 3 + 0] = dx * inv;
    s->xn[tid * 3 + 1] = dy * inv;
    s->xn[tid * 3 + 2] = dz * inv;
  }
  for (int idx = tid; idx < NH * 16 * 32; idx += 256) {
    int h = idx >> 9, rem = idx & 511, i = rem >> 5, dd = rem & 31;
    s->q[idx]    = (i < NCC) ? Qh[(row * NCC + i) * HID + h * HD + dd] : (_Float16)0.f;
    s->kmat[idx] = (i < NCC) ? Kh[(col * NCC + i) * HID + h * HD + dd] : (_Float16)0.f;
    s->attn16[idx] = (_Float16)0.f;
  }
  for (int idx = tid; idx < NH * 32 * 32; idx += 256) {
    int h = idx >> 10, rem = idx & 1023, dd = rem >> 5, j = rem & 31;
    s->vt[idx] = (j < NCC) ? Vh[(col * NCC + j) * HID + h * HD + dd] : (_Float16)0.f;
  }
  __syncthreads();

  // ---- P1: RBF features dist_rep (f16)
  const float step = 10.f / 63.f;
  const float coeff = -0.5f / (step * step);
  for (int idx = tid; idx < 208 * EC; idx += 256) {
    int ij = idx >> 6, c = idx & 63;
    float v = 0.f;
    if (ij < 196) { float dlt = s->rr[ij] - step * (float)c; v = __expf(coeff * dlt * dlt); }
    s->dist[idx] = (_Float16)v;
  }
  __syncthreads();

  // ---- P2: hidden = silu(dist @ sD_w1 + b1)   (13 M-tiles x 8 N-tiles, K=64)
  for (int job = wv; job < 104; job += 8) {
    int mt = job >> 3, nt = job & 7;
    v8f acc = {};
#pragma unroll
    for (int kk = 0; kk < 2; ++kk)
      acc = wmma16(load_a16(s->dist, EC, mt * 16, kk * 32, lane),
                   load_b16(s->w1t, EC, nt * 16, kk * 32, lane), acc);
    int crow = mt * 16 + ((lane & 16) ? 8 : 0), ccol = nt * 16 + (lane & 15);
    float bb = s->b1[ccol];
#pragma unroll
    for (int r = 0; r < 8; ++r)
      s->work[(crow + r) * HID + ccol] = (_Float16)silu_f(acc[r] + bb);
  }
  __syncthreads();

  // ---- P3: db = hidden @ sD_w2(pad 16) + b2
  for (int job = wv; job < 13; job += 8) {
    v8f acc = {};
#pragma unroll
    for (int kk = 0; kk < 4; ++kk)
      acc = wmma16(load_a16(s->work, HID, job * 16, kk * 32, lane),
                   load_b16(s->w2t, HID, 0, kk * 32, lane), acc);
    int crow = job * 16 + ((lane & 16) ? 8 : 0), ccol = lane & 15;
    if (ccol < 4) {
      float bb = s->b2[ccol];
#pragma unroll
      for (int r = 0; r < 8; ++r) s->db[(crow + r) * 4 + ccol] = acc[r] + bb;
    }
  }
  __syncthreads();

  // ---- P4: dv = dist @ sv_w + sv_b (reuses work); per-head logits
  for (int job = wv; job < 104; job += 8) {
    int mt = job >> 3, nt = job & 7;
    v8f acc = {};
#pragma unroll
    for (int kk = 0; kk < 2; ++kk)
      acc = wmma16(load_a16(s->dist, EC, mt * 16, kk * 32, lane),
                   load_b16(s->wvt, EC, nt * 16, kk * 32, lane), acc);
    int crow = mt * 16 + ((lane & 16) ? 8 : 0), ccol = nt * 16 + (lane & 15);
    float bb = s->svb[ccol];
#pragma unroll
    for (int r = 0; r < 8; ++r)
      s->work[(crow + r) * HID + ccol] = (_Float16)(acc[r] + bb);
  }
  if (wv < NH) {
    int h = wv;
    v8f acc = {};
    acc = wmma16(load_a16(s->q + h * 512, 32, 0, 0, lane),
                 load_b16(s->kmat + h * 512, 32, 0, 0, lane), acc);
    int crow = (lane & 16) ? 8 : 0, ccol = lane & 15;
#pragma unroll
    for (int r = 0; r < 8; ++r) {
      int i = crow + r, j = ccol;
      float v = -100000.f;
      if (i < NCC && j < NCC) {
        float ams = s->amr[i] * s->amc[j];
        if (ams > 0.f) v = acc[r] * 0.17677669529663687f + s->db[(i * NCC + j) * 4 + h];
      }
      s->logit[h * 256 + i * 16 + j] = v;
    }
  }
  __syncthreads();

  // ---- P5: per-row top-k(3), sparse mask, softmax + L1 renorm, r_ij partials
  if (tid < NH * NCC) {
    int h = tid / NCC, i = tid % NCC;
    float vals[NCC];
#pragma unroll
    for (int j = 0; j < NCC; ++j) vals[j] = s->logit[h * 256 + i * 16 + j];
    int sel0 = 0, sel1 = 0, sel2 = 0;
    float bv = -1e30f;
    for (int j = 0; j < NCC; ++j) if (vals[j] > bv) { bv = vals[j]; sel0 = j; }
    bv = -1e30f;
    for (int j = 0; j < NCC; ++j) if (j != sel0 && vals[j] > bv) { bv = vals[j]; sel1 = j; }
    bv = -1e30f;
    for (int j = 0; j < NCC; ++j) if (j != sel0 && j != sel1 && vals[j] > bv) { bv = vals[j]; sel2 = j; }
    s->topi[tid * SKK + 0] = sel0; s->topi[tid * SKK + 1] = sel1; s->topi[tid * SKK + 2] = sel2;
    float lp[NCC], m = -1e30f;
    for (int j = 0; j < NCC; ++j) {
      float sp = (j == sel0 || j == sel1 || j == sel2) ? 1.f : 0.f;
      float v = vals[j] * sp * (s->amr[i] * s->amc[j]);
      lp[j] = v; if (v > m) m = v;
    }
    float S = 0.f;
    for (int j = 0; j < NCC; ++j) S += __expf(lp[j] - m);
    float at[NCC], den2 = 0.f, num = 0.f, dcnt = 0.f;
    for (int j = 0; j < NCC; ++j) {
      float sp = (j == sel0 || j == sel1 || j == sel2) ? 1.f : 0.f;
      float ams = s->amr[i] * s->amc[j];
      float a = __expf(lp[j] - m) / S * ams * sp;
      at[j] = a; den2 += fabsf(a);
      num += lp[j]; dcnt += ams * sp;
    }
    den2 += 1e-7f;
    for (int j = 0; j < NCC; ++j) {
      float a = at[j] / den2;
      s->attnf[h * 256 + i * 16 + j] = a;
      s->attn16[h * 512 + i * 32 + j] = (_Float16)a;
    }
    s->rowN[h * 16 + i] = num; s->rowD[h * 16 + i] = dcnt;
  }
  __syncthreads();

  // ---- P6: r_ij to ws;  aV = attn @ V (per head, 2 N-tiles)
  if (tid < NH) {
    float num = 0.f, den = 0.f;
    for (int i = 0; i < NCC; ++i) { num += s->rowN[tid * 16 + i]; den += s->rowD[tid * 16 + i]; }
    rij_ws[e * NH + tid] = num / den;
  }
  for (int job = wv; job < 8; job += 8) {
    int h = job >> 1, nt = job & 1;
    v8f acc = {};
    acc = wmma16(load_a16(s->attn16 + h * 512, 32, 0, 0, lane),
                 load_b16(s->vt + h * 1024, 32, nt * 16, 0, lane), acc);
    int crow = (lane & 16) ? 8 : 0, ccol = lane & 15;
#pragma unroll
    for (int r = 0; r < 8; ++r)
      s->av16[h * 512 + (crow + r) * 32 + nt * 16 + ccol] = (_Float16)acc[r];
  }
  __syncthreads();

  // ---- P7: bmi layer1 (silu) -> h2;  triangle input gather -> tin
  for (int job = wv; job < 8; job += 8) {
    int h = job >> 1, nt = job & 1;
    v8f acc = {};
    acc = wmma16(load_a16(s->av16 + h * 512, 32, 0, 0, lane),
                 load_b16(s->bmi1t, 32, nt * 16, 0, lane), acc);
    int crow = (lane & 16) ? 8 : 0, ccol = nt * 16 + (lane & 15);
    float bb = s->bmib1[ccol];
#pragma unroll
    for (int r = 0; r < 8; ++r)
      s->h2[h * 512 + (crow + r) * 32 + ccol] = (_Float16)silu_f(acc[r] + bb);
  }
  for (int idx = tid; idx < 176 * 96; idx += 256) {
    int r2 = idx / 96, c = idx % 96;
    _Float16 v = (_Float16)0.f;
    if (r2 < 168) {
      int h = r2 / 42, rem = r2 % 42, i = rem / 3, s2 = rem % 3;
      int jg = s->topi[(h * NCC + i) * SKK + s2];
      if (c < 32)      v = s->q[h * 512 + i * 32 + c];
      else if (c < 64) v = s->kmat[h * 512 + jg * 32 + (c - 32)];
      else             v = s->work[(i * NCC + jg) * HID + h * HD + (c - 64)];
    }
    s->tin[idx] = v;
  }
  __syncthreads();

  // ---- P8: bmi layer2 -> ws aVm;  triangle GEMM1 -> thid
  for (int job = wv; job < 30; job += 8) {
    if (job < 8) {
      int h = job >> 1, nt = job & 1;
      v8f acc = {};
      acc = wmma16(load_a16(s->h2 + h * 512, 32, 0, 0, lane),
                   load_b16(s->bmi2t, 32, nt * 16, 0, lane), acc);
      int crow = (lane & 16) ? 8 : 0, ccol = nt * 16 + (lane & 15);
      float bb = s->bmib2[ccol];
#pragma unroll
      for (int r = 0; r < 8; ++r) {
        int i = crow + r;
        if (i < NCC)
          avm_ws[(((size_t)e * NH + h) * NCC + i) * HD + ccol] = acc[r] + bb;
      }
    } else {
      int j2 = job - 8, mt = j2 >> 1, nt = j2 & 1;
      v8f acc = {};
#pragma unroll
      for (int kk = 0; kk < 3; ++kk)
        acc = wmma16(load_a16(s->tin, 96, mt * 16, kk * 32, lane),
                     load_b16(s->tiw1t, 96, nt * 16, kk * 32, lane), acc);
      int crow = mt * 16 + ((lane & 16) ? 8 : 0), ccol = nt * 16 + (lane & 15);
      float bb = s->tib1[ccol];
#pragma unroll
      for (int r = 0; r < 8; ++r)
        s->thid[(crow + r) * 32 + ccol] = (_Float16)silu_f(acc[r] + bb);
    }
  }
  __syncthreads();

  // ---- P9: f = thid @ Ti_w2 + b2
  for (int t2 = tid; t2 < 168; t2 += 256) {
    float acc = s->tib2v;
#pragma unroll
    for (int k2 = 0; k2 < 32; ++k2) acc += (float)s->thid[t2 * 32 + k2] * s->tiw2[k2];
    s->fbuf[t2] = acc;
  }
  __syncthreads();

  // ---- P10: fv = sum_s (f * ag) * Xn_masked  -> ws
  if (tid < NH * NCC) {
    int h = tid / NCC, i = tid % NCC;
    float fx = 0.f, fy = 0.f, fz = 0.f;
#pragma unroll
    for (int s2 = 0; s2 < SKK; ++s2) {
      int jg = s->topi[tid * SKK + s2];
      float fac = s->fbuf[tid * 3 + s2] * s->attnf[h * 256 + i * 16 + jg] *
                  (s->amr[i] * s->amc[jg]);
      fx += fac * s->xn[(i * NCC + jg) * 3 + 0];
      fy += fac * s->xn[(i * NCC + jg) * 3 + 1];
      fz += fac * s->xn[(i * NCC + jg) * 3 + 2];
    }
    size_t o = (((size_t)e * NH + h) * NCC + i) * 3;
    fv_ws[o + 0] = fx; fv_ws[o + 1] = fy; fv_ws[o + 2] = fz;
  }
}

// ---------------- Kernel 3: per-node segment softmax + output projection ----------------
struct NodeSmem {
  alignas(16) _Float16 wot[HID * HID];   // WO_w^T [n][k]
  alignas(16) _Float16 upd[16 * HID];
  float beta[9 * NH];
  float wob[HID];
};

__global__ void node_kernel(const float* __restrict__ res_H, const float* __restrict__ res_X,
                            const float* __restrict__ atom_mask,
                            const unsigned char* __restrict__ residue_mask,
                            const float* __restrict__ WO_w, const float* __restrict__ WO_b,
                            const float* __restrict__ rij, const float* __restrict__ avm,
                            const float* __restrict__ fvv,
                            float* __restrict__ outH, float* __restrict__ outX) {
  __shared__ NodeSmem sm;
  const int n = blockIdx.x, tid = threadIdx.x, wv = tid >> 5, lane = tid & 31;

  for (int idx = tid; idx < HID * HID; idx += 256) {
    int nW = idx >> 7, k = idx & 127;
    sm.wot[idx] = (_Float16)WO_w[k * HID + nW];
  }
  for (int idx = tid; idx < HID; idx += 256) sm.wob[idx] = WO_b[idx];
  if (tid < NH) {   // edges of node n: n*8+k (k<8) and NRES*8+n (self)
    float rv[9]; float m = -1e30f;
    for (int k = 0; k < 9; ++k) {
      int e = (k < 8) ? (n * 8 + k) : (NRES * 8 + n);
      rv[k] = rij[e * NH + tid]; m = fmaxf(m, rv[k]);
    }
    float S = 0.f;
    for (int k = 0; k < 9; ++k) { rv[k] = __expf(rv[k] - m); S += rv[k]; }
    for (int k = 0; k < 9; ++k) sm.beta[k * NH + tid] = rv[k] / S;
  }
  __syncthreads();

  for (int idx = tid; idx < 16 * HID; idx += 256) {
    int i = idx >> 7, c = idx & 127;
    _Float16 v = (_Float16)0.f;
    if (i < NCC) {
      int h = c >> 5, dd = c & 31;
      float acc = 0.f;
      for (int k = 0; k < 9; ++k) {
        int e = (k < 8) ? (n * 8 + k) : (NRES * 8 + n);
        acc += sm.beta[k * NH + h] * avm[(((size_t)e * NH + h) * NCC + i) * HD + dd];
      }
      v = (_Float16)acc;
    }
    sm.upd[idx] = v;
  }
  __syncthreads();

  for (int job = wv; job < 8; job += 8) {
    v8f acc = {};
#pragma unroll
    for (int kk = 0; kk < 4; ++kk)
      acc = wmma16(load_a16(sm.upd, HID, 0, kk * 32, lane),
                   load_b16(sm.wot, HID, job * 16, kk * 32, lane), acc);
    int crow = (lane & 16) ? 8 : 0, ccol = job * 16 + (lane & 15);
#pragma unroll
    for (int r = 0; r < 8; ++r) {
      int i = crow + r;
      if (i < NCC) {
        size_t gi = ((size_t)n * NCC + i) * HID + ccol;
        outH[gi] = (res_H[gi] + acc[r] + sm.wob[ccol]) * atom_mask[n * NCC + i];
      }
    }
  }
  if (tid < NCC) {
    int i = tid;
    float am = atom_mask[n * NCC + i];
    bool rm = residue_mask[n] != 0;
    for (int x = 0; x < 3; ++x) {
      float acc = 0.f;
      for (int h = 0; h < NH; ++h)
        for (int k = 0; k < 9; ++k) {
          int e = (k < 8) ? (n * 8 + k) : (NRES * 8 + n);
          acc += sm.beta[k * NH + h] * fvv[(((size_t)e * NH + h) * NCC + i) * 3 + x];
        }
      acc = fminf(fmaxf(acc, -3.f), 3.f);
      float base = res_X[(n * NCC + i) * 3 + x];
      outX[(n * NCC + i) * 3 + x] = (rm ? base + acc : base) * am;
    }
  }
}

// ---------------- Host launcher ----------------
extern "C" void kernel_launch(void* const* d_in, const int* in_sizes, int n_in,
                              void* d_out, int out_size, void* d_ws, size_t ws_size,
                              hipStream_t stream) {
  (void)in_sizes; (void)n_in; (void)out_size; (void)ws_size;
  const float* res_H        = (const float*)d_in[0];
  const float* res_X        = (const float*)d_in[1];
  const float* atom_mask    = (const float*)d_in[2];
  const unsigned char* rmsk = (const unsigned char*)d_in[3];
  const int*   edge_index   = (const int*)d_in[4];
  const float* WQ     = (const float*)d_in[5];
  const float* WK     = (const float*)d_in[6];
  const float* WV     = (const float*)d_in[7];
  const float* WO_w   = (const float*)d_in[8];
  const float* WO_b   = (const float*)d_in[9];
  const float* Ti_w1  = (const float*)d_in[10];
  const float* Ti_b1  = (const float*)d_in[11];
  const float* Ti_w2  = (const float*)d_in[12];
  const float* Ti_b2  = (const float*)d_in[13];
  const float* sD_w1  = (const float*)d_in[14];
  const float* sD_b1  = (const float*)d_in[15];
  const float* sD_w2  = (const float*)d_in[16];
  const float* sD_b2  = (const float*)d_in[17];
  const float* sv_w   = (const float*)d_in[18];
  const float* sv_b   = (const float*)d_in[19];
  const float* bmi_w1 = (const float*)d_in[20];
  const float* bmi_b1 = (const float*)d_in[21];
  const float* bmi_w2 = (const float*)d_in[22];
  const float* bmi_b2 = (const float*)d_in[23];

  char* ws = (char*)d_ws;
  const size_t QBYTES = (size_t)MROW * HID * sizeof(_Float16);   // 2,752,512
  _Float16* Qh  = (_Float16*)(ws);
  _Float16* Kh  = (_Float16*)(ws + QBYTES);
  _Float16* Vh  = (_Float16*)(ws + 2 * QBYTES);
  float*    rij = (float*)(ws + 3 * QBYTES);
  float*    avm = (float*)(ws + 3 * QBYTES + (size_t)NEDGE * NH * sizeof(float));
  float*    fvv = (float*)(ws + 3 * QBYTES + (size_t)NEDGE * NH * sizeof(float)
                              + (size_t)NEDGE * NH * NCC * HD * sizeof(float));

  float* outH = (float*)d_out;
  float* outX = outH + (size_t)MROW * HID;

  qkv_kernel<<<MROW / 16, 256, sizeof(QkvSmem), stream>>>(res_H, WQ, WK, WV, Qh, Kh, Vh);
  edge_kernel<<<NEDGE, 256, sizeof(EdgeSmem), stream>>>(
      res_X, atom_mask, edge_index,
      sD_w1, sD_b1, sD_w2, sD_b2, sv_w, sv_b,
      Ti_w1, Ti_b1, Ti_w2, Ti_b2, bmi_w1, bmi_b1, bmi_w2, bmi_b2,
      Qh, Kh, Vh, rij, avm, fvv);
  node_kernel<<<NRES, 256, 0, stream>>>(res_H, res_X, atom_mask, rmsk,
                                        WO_w, WO_b, rij, avm, fvv, outH, outX);
}
